// SparseConvBlock_38843684225423
// MI455X (gfx1250) — compile-verified, matrix operand order checked by reading, not run on hardware
//
#include <hip/hip_runtime.h>

// CDNA5 / gfx1250 sparse 3D conv (gather-GEMM-scatter) + BatchNorm + ReLU.
// GEMM path: v_wmma_f32_16x16x32_f16, wave32, A gathered straight into
// fragment layout, B staged transposed in LDS as f16.

typedef __attribute__((ext_vector_type(16))) _Float16 v16h;
typedef __attribute__((ext_vector_type(8)))  _Float16 v8h;
typedef __attribute__((ext_vector_type(8)))  float    v8f;
typedef __attribute__((ext_vector_type(4)))  float    v4f;

#define CIN        128
#define COUT       128
#define NKERN      27
#define LDS_PITCH  136   // f16 elems per LDS row: 272B -> 4-bank rotation/row, 16B aligned

// ---------------------------------------------------------------- zero init
__global__ void zero_kernel(float* __restrict__ out, float* __restrict__ sums,
                            long long n) {
  long long i = (long long)blockIdx.x * blockDim.x + threadIdx.x;
  long long stride = (long long)gridDim.x * blockDim.x;
  for (; i < n; i += stride) out[i] = 0.0f;
  if (blockIdx.x == 0 && threadIdx.x < 2 * COUT) sums[threadIdx.x] = 0.0f;
}

// ------------------------------------------------- gather - WMMA - scatter
// grid = ( ceil(M/256), 27 ), block = 128 (4 waves); each wave: 16x128 tile.
__global__ void __launch_bounds__(128)
conv_kernel(const float* __restrict__ feats, const float* __restrict__ W,
            const int* __restrict__ in_map, const int* __restrict__ out_map,
            float* __restrict__ acc_out, int M) {
  __shared__ _Float16 wt[COUT * LDS_PITCH];   // W[k] transposed: wt[c_out][c_in]

  const int k    = blockIdx.y;
  const int tid  = threadIdx.x;
  const int lane = tid & 31;
  const int wave = tid >> 5;
  const int half = lane >> 4;      // 0: K 0-15, 1: K 16-31 within a 32-chunk
  const int l16  = lane & 15;

  // ---- stage W[k] (f32, [c_in][c_out]) into LDS transposed as f16 ----
  const float* wk = W + (long long)k * (CIN * COUT);
  for (int i = tid * 4; i < CIN * COUT; i += 128 * 4) {
    v4f w4 = *(const v4f*)(wk + i);           // 4 consecutive c_out, same c_in
    int cin  = i >> 7;
    int cout = i & 127;
    wt[(cout + 0) * LDS_PITCH + cin] = (_Float16)w4[0];
    wt[(cout + 1) * LDS_PITCH + cin] = (_Float16)w4[1];
    wt[(cout + 2) * LDS_PITCH + cin] = (_Float16)w4[2];
    wt[(cout + 3) * LDS_PITCH + cin] = (_Float16)w4[3];
  }
  __syncthreads();

  const int mbase = blockIdx.x * 256;

  for (int iter = 0; iter < 4; ++iter) {
    const int m0  = mbase + iter * 64 + wave * 16;   // 16 messages per wave
    const int ml  = m0 + l16;
    const int mcl = (ml < M) ? ml : (M - 1);         // clamp (tail predicated at scatter)
    const int iidx = in_map[k * M + mcl];

    // A-fragment source: lane holds 16 contiguous f32 of its gathered row
    const float* arow = feats + (long long)iidx * CIN + (half << 4);

    v8f acc[8] = {};   // 16x128 f32 accumulator tile (8 x 16x16)

#pragma unroll
    for (int kk = 0; kk < 4; ++kk) {         // K = 128 reduction, 32 per WMMA
      const float* ap = arow + kk * 32;
      v4f a0 = *(const v4f*)(ap + 0);
      v4f a1 = *(const v4f*)(ap + 4);
      v4f a2 = *(const v4f*)(ap + 8);
      v4f a3 = *(const v4f*)(ap + 12);
      v16h a;
#pragma unroll
      for (int t = 0; t < 4; ++t) {
        a[t]      = (_Float16)a0[t];
        a[4 + t]  = (_Float16)a1[t];
        a[8 + t]  = (_Float16)a2[t];
        a[12 + t] = (_Float16)a3[t];
      }

#pragma unroll
      for (int nt = 0; nt < 8; ++nt) {       // 8 N-tiles of 16 columns
        const _Float16* bp =
            &wt[(nt * 16 + l16) * LDS_PITCH + kk * 32 + (half << 4)];
        v8h b0 = *(const v8h*)(bp);
        v8h b1 = *(const v8h*)(bp + 8);
        v16h b = __builtin_shufflevector(b0, b1, 0, 1, 2, 3, 4, 5, 6, 7,
                                         8, 9, 10, 11, 12, 13, 14, 15);
        acc[nt] = __builtin_amdgcn_wmma_f32_16x16x32_f16(
            false, a, false, b, (short)0, acc[nt], false, false);
      }
    }

    // ---- scatter-add: row M=j (lanes 0-15) / j+8 (lanes 16-31), col=l16+16*nt
    const int oidx_l = out_map[k * M + mcl];
#pragma unroll
    for (int j = 0; j < 8; ++j) {
      const int srcl = (lane < 16) ? j : (j + 8);
      const int oidx = __shfl(oidx_l, srcl, 32);
      const int mrow = m0 + srcl;
      if (mrow < M) {
        float* dst = acc_out + (long long)oidx * COUT + l16;
#pragma unroll
        for (int nt = 0; nt < 8; ++nt)
          atomicAdd(dst + nt * 16, acc[nt][j]);   // 64B contiguous segments
      }
    }
  }
}

// ------------------------------------------------- BN stats (sum, sumsq)
__global__ void reduce_kernel(const float* __restrict__ x,
                              float* __restrict__ sums, int N) {
  const int c    = threadIdx.x & 127;
  const int rsub = threadIdx.x >> 7;                 // 0..1
  float s = 0.0f, s2 = 0.0f;
  for (long long r = (long long)blockIdx.x * 2 + rsub; r < N;
       r += (long long)gridDim.x * 2) {
    float v = x[r * COUT + c];
    s += v;
    s2 += v * v;
  }
  atomicAdd(&sums[c], s);
  atomicAdd(&sums[COUT + c], s2);
}

// ------------------------------------------------- normalize + ReLU (in place)
__global__ void bn_relu_kernel(float* __restrict__ out,
                               const float* __restrict__ sums,
                               const float* __restrict__ gamma,
                               const float* __restrict__ beta, int N) {
  long long i = (long long)blockIdx.x * blockDim.x + threadIdx.x;
  long long total = (long long)N * COUT;
  if (i >= total) return;
  const int c = (int)(i & 127);
  const float inv_n = 1.0f / (float)N;
  const float mean  = sums[c] * inv_n;
  const float var   = sums[COUT + c] * inv_n - mean * mean;
  const float scale = gamma[c] * __frsqrt_rn(var + 1e-5f);
  const float y = (out[i] - mean) * scale + beta[c];
  out[i] = fmaxf(y, 0.0f);
}

extern "C" void kernel_launch(void* const* d_in, const int* in_sizes, int n_in,
                              void* d_out, int out_size, void* d_ws,
                              size_t ws_size, hipStream_t stream) {
  const float* feats  = (const float*)d_in[0];
  const float* W      = (const float*)d_in[1];
  const float* gamma  = (const float*)d_in[2];
  const float* beta   = (const float*)d_in[3];
  const int*   in_map = (const int*)d_in[4];
  const int*   out_map= (const int*)d_in[5];
  float* out  = (float*)d_out;
  float* sums = (float*)d_ws;                 // 256 floats: [sum | sumsq]

  const int N = in_sizes[0] / CIN;            // 200000
  const int M = in_sizes[4] / NKERN;          // 100000
  const long long total = (long long)N * COUT;

  zero_kernel<<<2048, 256, 0, stream>>>(out, sums, total);

  dim3 grid((M + 255) / 256, NKERN);
  conv_kernel<<<grid, 128, 0, stream>>>(feats, W, in_map, out_map, out, M);

  reduce_kernel<<<1024, 256, 0, stream>>>(out, sums, N);

  int blocks = (int)((total + 255) / 256);
  bn_relu_kernel<<<blocks, 256, 0, stream>>>(out, sums, gamma, beta, N);
}
